// MoEGate_3006477107309
// MI455X (gfx1250) — matrix operand order repaired and checked by reading, not compiled, not run on hardware
//
#include <hip/hip_runtime.h>
#include <hip/hip_bf16.h>

typedef __bf16 bf16;
typedef __attribute__((ext_vector_type(16))) __bf16 v16bf;
typedef __attribute__((ext_vector_type(8)))  __bf16 v8bf;
typedef __attribute__((ext_vector_type(8)))  float  v8f;
typedef __attribute__((ext_vector_type(4)))  float  v4f;

#define N_EXPERTS   256
#define N_GROUP     8
#define E_PER_G     32
#define TOPK_GROUP  4
#define TOP_K       8
#define HDIM        7168
#define KC          64      // K-chunk staged in LDS per iteration
#define RSCALE      2.5f

__device__ __forceinline__ v16bf cat8(v8bf lo, v8bf hi) {
    return __builtin_shufflevector(lo, hi, 0,1,2,3,4,5,6,7, 8,9,10,11,12,13,14,15);
}
__device__ __forceinline__ float sigmoidf_(float x) {
    return 1.0f / (1.0f + __expf(-x));
}

// ---------------------------------------------------------------------------
// Pre-pass: W [HDIM,256] fp32 row-major  ->  W_t [256,HDIM] bf16 (expert-major)
// ---------------------------------------------------------------------------
__global__ __launch_bounds__(256) void w_transpose_kernel(
    const float* __restrict__ w, bf16* __restrict__ wt)
{
    __shared__ float tile[32][33];              // [k][e], padded vs bank conflicts
    const int e0 = blockIdx.x * 32;
    const int k0 = blockIdx.y * 32;
    const int te = threadIdx.x & 31;            // 0..31
    const int tk = threadIdx.x >> 5;            // 0..7
    #pragma unroll
    for (int i = 0; i < 32; i += 8)             // coalesced in e across lanes
        tile[tk + i][te] = w[(size_t)(k0 + tk + i) * N_EXPERTS + (e0 + te)];
    __syncthreads();
    #pragma unroll
    for (int i = 0; i < 32; i += 8)             // coalesced in k across lanes
        wt[(size_t)(e0 + tk + i) * HDIM + (k0 + te)] = (bf16)tile[te][tk + i];
}

// ---------------------------------------------------------------------------
// Main: 32 tokens x 256 experts per workgroup. X via LDS (shared across waves),
// B fragments straight from L2-resident bf16 W_t (no intra-WG redundancy).
// ---------------------------------------------------------------------------
__global__ __launch_bounds__(256) void moe_gate_kernel(
    const float* __restrict__ x,      // [n_tok, HDIM]
    const bf16*  __restrict__ wt,     // [256, HDIM] bf16 transposed
    const float* __restrict__ bias,   // [256]
    float* __restrict__ out_idx,      // [n_tok, 8] (indices as float)
    float* __restrict__ out_w)        // [n_tok, 8]
{
    constexpr int M_TILE = 32;
    __shared__ bf16  lds_x[M_TILE * KC];            // 4 KB
    __shared__ float lds_s[M_TILE * N_EXPERTS];     // 32 KB
    __shared__ float lds_gs[M_TILE * N_GROUP];      // 1 KB

    const int tid  = threadIdx.x;
    const int lane = tid & 31;
    const int wave = tid >> 5;
    const int tok0 = blockIdx.x * M_TILE;
    const int e_base = wave * 32;                   // 32 experts per wave

    v8f acc00 = {}, acc01 = {}, acc10 = {}, acc11 = {};

    // 16-bit A (16x32), contiguous k order chosen here: lane m = lane&15; lanes 0-15 hold K 0..15,
    // lanes 16-31 hold K 16..31 (matches B).
    const int a_m  = lane & 15;
    const int a_kb = (lane < 16) ? 0 : 16;
    // 16-bit B (32x16), same contiguous k order as A: column n = lane&15; lanes 0-15 K=0..15, 16-31 K=16..31.
    const int b_n  = lane & 15;
    const int b_kb = (lane < 16) ? 0 : 16;

    // X staging mapping: thread -> (token, 8 consecutive k)
    const int sx_m  = tid >> 3;                     // 0..31
    const int sx_k  = (tid & 7) << 3;               // 0..56 step 8

    for (int k0 = 0; k0 < HDIM; k0 += KC) {
        // ---- stage X tile: 32 x 64 fp32 -> bf16 LDS [m][k] (one b128 store) ----
        {
            const float* xp = x + (size_t)(tok0 + sx_m) * HDIM + k0 + sx_k;
            v4f a = *(const v4f*)(xp);
            v4f b = *(const v4f*)(xp + 4);
            v8bf p;
            p[0]=(bf16)a.x; p[1]=(bf16)a.y; p[2]=(bf16)a.z; p[3]=(bf16)a.w;
            p[4]=(bf16)b.x; p[5]=(bf16)b.y; p[6]=(bf16)b.z; p[7]=(bf16)b.w;
            *(v8bf*)(lds_x + sx_m * KC + sx_k) = p;
            if (k0 + KC < HDIM)                     // global_prefetch next X chunk
                __builtin_prefetch(xp + KC, 0, 1);
        }
        __syncthreads();

        // ---- compute: 2 K-steps of 32; 2 M-tiles x 2 N-tiles -> 4 WMMA ----
        #pragma unroll
        for (int ks = 0; ks < KC; ks += 32) {
            const bf16* xr0 = lds_x + a_m * KC + ks + a_kb;          // tokens 0-15
            const bf16* xr1 = xr0 + 16 * KC;                          // tokens 16-31
            v16bf a0 = cat8(*(const v8bf*)xr0, *(const v8bf*)(xr0 + 8));
            v16bf a1 = cat8(*(const v8bf*)xr1, *(const v8bf*)(xr1 + 8));

            const bf16* wp = wt + (size_t)(e_base + b_n) * HDIM + k0 + ks + b_kb;
            v16bf b0 = *(const v16bf*)(wp);                 // experts e_base..+15
            v16bf b1 = *(const v16bf*)(wp + 16 * HDIM);     // experts e_base+16..+31

            acc00 = __builtin_amdgcn_wmma_f32_16x16x32_bf16(
                        false, a0, false, b0, (short)0, acc00, false, false);
            acc01 = __builtin_amdgcn_wmma_f32_16x16x32_bf16(
                        false, a0, false, b1, (short)0, acc01, false, false);
            acc10 = __builtin_amdgcn_wmma_f32_16x16x32_bf16(
                        false, a1, false, b0, (short)0, acc10, false, false);
            acc11 = __builtin_amdgcn_wmma_f32_16x16x32_bf16(
                        false, a1, false, b1, (short)0, acc11, false, false);
        }
        __syncthreads();
    }

    // ---- epilogue: sigmoid + bias -> scores in LDS ----
    // C/D layout: VGPR j -> row M=j (lanes 0-15) / M=j+8 (lanes 16-31), N=lane&15.
    {
        const int mb = (lane >= 16) ? 8 : 0;
        const int n0 = e_base + (lane & 15);
        const int n1 = n0 + 16;
        const float bia0 = bias[n0], bia1 = bias[n1];
        #pragma unroll
        for (int j = 0; j < 8; ++j) {
            lds_s[(mb + j) * N_EXPERTS + n0]        = sigmoidf_(acc00[j]) + bia0;
            lds_s[(mb + j) * N_EXPERTS + n1]        = sigmoidf_(acc01[j]) + bia1;
            lds_s[(16 + mb + j) * N_EXPERTS + n0]   = sigmoidf_(acc10[j]) + bia0;
            lds_s[(16 + mb + j) * N_EXPERTS + n1]   = sigmoidf_(acc11[j]) + bia1;
        }
    }
    __syncthreads();

    // ---- group scores: 32 tokens x 8 groups == 256 threads exactly ----
    {
        const int m = tid >> 3, g = tid & 7;
        const float* s = lds_s + m * N_EXPERTS + g * E_PER_G;
        float m1 = -INFINITY, m2 = -INFINITY;
        for (int i = 0; i < E_PER_G; ++i) {
            float v = s[i];
            if (v > m1) { m2 = m1; m1 = v; }
            else if (v > m2) { m2 = v; }
        }
        lds_gs[m * N_GROUP + g] = m1 + m2;
    }
    __syncthreads();

    // ---- per-token: top-4 groups, masked top-8 experts, normalize ----
    if (tid < M_TILE) {
        const int m = tid;
        const float* gs = lds_gs + m * N_GROUP;

        unsigned gmask = 0;
        for (int r = 0; r < TOPK_GROUP; ++r) {
            float best = -INFINITY; int bi = 0;
            for (int g = 0; g < N_GROUP; ++g)
                if (!((gmask >> g) & 1u) && gs[g] > best) { best = gs[g]; bi = g; }
            gmask |= 1u << bi;
        }

        const float* s = lds_s + m * N_EXPERTS;
        int idx[TOP_K]; float wv[TOP_K];
        unsigned used[N_EXPERTS / 32] = {0,0,0,0,0,0,0,0};
        float sum = 0.0f;
        for (int r = 0; r < TOP_K; ++r) {
            float best = -INFINITY; int bi = 0;
            for (int e = 0; e < N_EXPERTS; ++e) {
                if ((used[e >> 5] >> (e & 31)) & 1u) continue;
                float v = ((gmask >> (e >> 5)) & 1u) ? s[e] : 0.0f; // ref masks w/ 0.0
                if (v > best) { best = v; bi = e; }
            }
            used[bi >> 5] |= 1u << (bi & 31);
            idx[r] = bi;
            wv[r]  = s[bi];
            sum   += wv[r];
        }
        const float inv = RSCALE / (sum + 1e-20f);
        const size_t tg = (size_t)(tok0 + m) * TOP_K;
        #pragma unroll
        for (int r = 0; r < TOP_K; ++r) {
            out_idx[tg + r] = (float)idx[r];
            out_w[tg + r]   = wv[r] * inv;
        }
    }
}

// ---------------------------------------------------------------------------
// Fallback (no workspace): in-kernel W staging, 16 tokens per WG (round-1 code)
// ---------------------------------------------------------------------------
__global__ __launch_bounds__(256) void moe_gate_fallback(
    const float* __restrict__ x, const float* __restrict__ w,
    const float* __restrict__ bias,
    float* __restrict__ out_idx, float* __restrict__ out_w)
{
    constexpr int M_TILE = 16;
    __shared__ bf16  lds_x[M_TILE * KC];
    __shared__ bf16  lds_w[N_EXPERTS * KC];
    __shared__ float lds_s[M_TILE * N_EXPERTS];
    __shared__ float lds_gs[M_TILE * N_GROUP];

    const int tid  = threadIdx.x;
    const int lane = tid & 31;
    const int wave = tid >> 5;
    const int tok0 = blockIdx.x * M_TILE;
    const int e_base = wave * 32;
    v8f acc0 = {}, acc1 = {};
    const int a_m  = lane & 15;
    const int a_kb = (lane < 16) ? 0 : 16;
    const int b_n  = lane & 15;
    const int b_kb = (lane < 16) ? 0 : 16;

    for (int k0 = 0; k0 < HDIM; k0 += KC) {
        {
            const int m  = tid >> 4;
            const int kk = (tid & 15) << 2;
            v4f v = *(const v4f*)(x + (size_t)(tok0 + m) * HDIM + k0 + kk);
            bf16* dst = lds_x + m * KC + kk;
            dst[0]=(bf16)v.x; dst[1]=(bf16)v.y; dst[2]=(bf16)v.z; dst[3]=(bf16)v.w;
        }
        {
            const float* wp = w + (size_t)k0 * N_EXPERTS + tid;
            bf16* dst = lds_w + tid * KC;
            #pragma unroll 8
            for (int k = 0; k < KC; k += 2) {
                dst[k]   = (bf16)wp[(size_t)(k)     * N_EXPERTS];
                dst[k+1] = (bf16)wp[(size_t)(k + 1) * N_EXPERTS];
            }
        }
        __syncthreads();
        #pragma unroll
        for (int ks = 0; ks < KC; ks += 32) {
            const bf16* xr = lds_x + a_m * KC + ks + a_kb;
            v16bf a = cat8(*(const v8bf*)xr, *(const v8bf*)(xr + 8));
            v16bf b0 = *(const v16bf*)(lds_w + (e_base      + b_n) * KC + ks + b_kb);
            v16bf b1 = *(const v16bf*)(lds_w + (e_base + 16 + b_n) * KC + ks + b_kb);
            acc0 = __builtin_amdgcn_wmma_f32_16x16x32_bf16(
                       false, a, false, b0, (short)0, acc0, false, false);
            acc1 = __builtin_amdgcn_wmma_f32_16x16x32_bf16(
                       false, a, false, b1, (short)0, acc1, false, false);
        }
        __syncthreads();
    }
    {
        const int mb = (lane >= 16) ? 8 : 0;
        const int n0 = e_base + (lane & 15);
        const int n1 = n0 + 16;
        #pragma unroll
        for (int j = 0; j < 8; ++j) {
            lds_s[(mb + j) * N_EXPERTS + n0] = sigmoidf_(acc0[j]) + bias[n0];
            lds_s[(mb + j) * N_EXPERTS + n1] = sigmoidf_(acc1[j]) + bias[n1];
        }
    }
    __syncthreads();
    if (tid < M_TILE * N_GROUP) {
        const int m = tid >> 3, g = tid & 7;
        const float* s = lds_s + m * N_EXPERTS + g * E_PER_G;
        float m1 = -INFINITY, m2 = -INFINITY;
        for (int i = 0; i < E_PER_G; ++i) {
            float v = s[i];
            if (v > m1) { m2 = m1; m1 = v; } else if (v > m2) { m2 = v; }
        }
        lds_gs[m * N_GROUP + g] = m1 + m2;
    }
    __syncthreads();
    if (tid < M_TILE) {
        const int m = tid;
        const float* gs = lds_gs + m * N_GROUP;
        unsigned gmask = 0;
        for (int r = 0; r < TOPK_GROUP; ++r) {
            float best = -INFINITY; int bi = 0;
            for (int g = 0; g < N_GROUP; ++g)
                if (!((gmask >> g) & 1u) && gs[g] > best) { best = gs[g]; bi = g; }
            gmask |= 1u << bi;
        }
        const float* s = lds_s + m * N_EXPERTS;
        int idx[TOP_K]; float wv[TOP_K];
        unsigned used[N_EXPERTS / 32] = {0,0,0,0,0,0,0,0};
        float sum = 0.0f;
        for (int r = 0; r < TOP_K; ++r) {
            float best = -INFINITY; int bi = 0;
            for (int e = 0; e < N_EXPERTS; ++e) {
                if ((used[e >> 5] >> (e & 31)) & 1u) continue;
                float v = ((gmask >> (e >> 5)) & 1u) ? s[e] : 0.0f;
                if (v > best) { best = v; bi = e; }
            }
            used[bi >> 5] |= 1u << (bi & 31);
            idx[r] = bi; wv[r] = s[bi]; sum += wv[r];
        }
        const float inv = RSCALE / (sum + 1e-20f);
        const size_t tg = (size_t)(tok0 + m) * TOP_K;
        #pragma unroll
        for (int r = 0; r < TOP_K; ++r) {
            out_idx[tg + r] = (float)idx[r];
            out_w[tg + r]   = wv[r] * inv;
        }
    }
}

extern "C" void kernel_launch(void* const* d_in, const int* in_sizes, int n_in,
                              void* d_out, int out_size, void* d_ws, size_t ws_size,
                              hipStream_t stream) {
    const float* x    = (const float*)d_in[0];  // hidden_states [4,4096,7168]
    const float* w    = (const float*)d_in[1];  // weight [7168,256]
    const float* bias = (const float*)d_in[2];  // bias [256]

    const int n_tok = in_sizes[0] / HDIM;       // 16384
    float* out     = (float*)d_out;
    float* out_idx = out;                        // first n_tok*8 elems
    float* out_w   = out + (size_t)n_tok * TOP_K;

    const size_t wt_bytes = (size_t)HDIM * N_EXPERTS * sizeof(bf16);  // 3.67 MB
    if (ws_size >= wt_bytes) {
        bf16* wt = (bf16*)d_ws;
        dim3 tgrid(N_EXPERTS / 32, HDIM / 32);   // 8 x 224 tiles
        w_transpose_kernel<<<tgrid, 256, 0, stream>>>(w, wt);
        moe_gate_kernel<<<dim3(n_tok / 32), 256, 0, stream>>>(
            x, wt, bias, out_idx, out_w);
    } else {
        moe_gate_fallback<<<dim3(n_tok / 16), 256, 0, stream>>>(
            x, w, bias, out_idx, out_w);
    }
}
